// BjorckLinear_2491081032277
// MI455X (gfx1250) — matrix operand order repaired and checked by reading, not compiled, not run on hardware
//
#include <hip/hip_runtime.h>
#include <hip/hip_bf16.h>

#define USE_ASYNC_B 1   // global_load_async_to_lds_b128 for the W-operand staging

typedef __attribute__((ext_vector_type(2)))  float  v2f;
typedef __attribute__((ext_vector_type(8)))  float  v8f;
typedef __attribute__((ext_vector_type(8)))  __bf16 v8bf;
typedef __attribute__((ext_vector_type(16))) __bf16 v16bf;

__device__ __forceinline__ unsigned short f2bf_rne(float f) {
    unsigned int u = __float_as_uint(f);
    unsigned int r = u + 0x7FFFu + ((u >> 16) & 1u);
    return (unsigned short)(r >> 16);
}

__device__ __forceinline__ void wait_async0() {
#if __has_builtin(__builtin_amdgcn_s_wait_asynccnt)
    __builtin_amdgcn_s_wait_asynccnt(0);
#else
    asm volatile("s_wait_asynccnt 0x0" ::: "memory");
#endif
}

// Async copy of 16 bytes global -> LDS (per lane). ASYNCcnt-tracked, no VGPR data.
__device__ __forceinline__ void async_copy_b128(const void* gptr, void* lds_ptr) {
#if USE_ASYNC_B
    unsigned int       lds = (unsigned int)(unsigned long long)(uintptr_t)lds_ptr;
    unsigned long long ga  = (unsigned long long)(uintptr_t)gptr;
    asm volatile("global_load_async_to_lds_b128 %0, %1, off"
                 :: "v"(lds), "v"(ga) : "memory");
#else
    *(uint4*)lds_ptr = *(const uint4*)gptr;
#endif
}

// ---------------------------------------------------------------------------
// Phase 0: scale kernel by 1/sqrt(1024*1024) = 1/1024
// ---------------------------------------------------------------------------
__global__ void scale_w(const float* __restrict__ in, float* __restrict__ out) {
    int i = blockIdx.x * blockDim.x + threadIdx.x;
    out[i] = in[i] * (1.0f / 1024.0f);
}

// ---------------------------------------------------------------------------
// Phase 1: Bjorck iteration GEMMs, fp32, V_WMMA_F32_16X16X4_F32.
//   TRANSA=false, UPDATE=false : Out = A^T * B           (WtW = W^T W)
//   TRANSA=true,  UPDATE=true  : Out = 1.5*Worig - 0.5*(A * B)
// 64x64 tile per 128-thread block (4 waves, each 32x32 = 2x2 WMMA tiles).
// ---------------------------------------------------------------------------
template<bool TRANSA, bool UPDATE>
__global__ __launch_bounds__(128) void bjorck_gemm(
        const float* __restrict__ A, const float* __restrict__ B,
        const float* __restrict__ Worig, float* __restrict__ Out) {
    __shared__ float Am[64][64];   // Am[k][m] : A-operand, K-major
    __shared__ float Bs[64][64];   // Bs[k][n]
    const int tid  = threadIdx.x;
    const int lane = tid & 31;
    const int wid  = tid >> 5;
    const int m0 = blockIdx.y * 64;
    const int n0 = blockIdx.x * 64;
    const int wm = (wid >> 1) * 32;
    const int wn = (wid & 1) * 32;
    const int l15 = lane & 15;
    const int sel = (lane < 16) ? 0 : 2;   // f32 A/B operand: lanes 16-31 hold K=2,3

    v8f acc[2][2] = {};

    for (int k0 = 0; k0 < 1024; k0 += 64) {
        __syncthreads();
        #pragma unroll
        for (int i = 0; i < 8; ++i) {
            int idx = tid + i * 128;            // 1024 float4 per 64x64 tile
            int r   = idx >> 4;
            int c4  = (idx & 15) * 4;
            if (TRANSA) {
                // Am[k][m] = A[m0+m][k0+k]  (transpose on store)
                float4 v = *(const float4*)(A + (long)(m0 + r) * 1024 + k0 + c4);
                Am[c4 + 0][r] = v.x;
                Am[c4 + 1][r] = v.y;
                Am[c4 + 2][r] = v.z;
                Am[c4 + 3][r] = v.w;
            } else {
                // Am[k][m] = A[k0+k][m0+m]  (direct copy)
                *(float4*)&Am[r][c4] = *(const float4*)(A + (long)(k0 + r) * 1024 + m0 + c4);
            }
            *(float4*)&Bs[r][c4] = *(const float4*)(B + (long)(k0 + r) * 1024 + n0 + c4);
        }
        __syncthreads();

        for (int kk = 0; kk < 64; kk += 4) {
#if __has_builtin(__builtin_amdgcn_wmma_f32_16x16x4_f32)
            v2f a0, a1, b0, b1;
            a0.x = Am[kk + sel][wm + l15];       a0.y = Am[kk + sel + 1][wm + l15];
            a1.x = Am[kk + sel][wm + 16 + l15];  a1.y = Am[kk + sel + 1][wm + 16 + l15];
            b0.x = Bs[kk + sel][wn + l15];       b0.y = Bs[kk + sel + 1][wn + l15];
            b1.x = Bs[kk + sel][wn + 16 + l15];  b1.y = Bs[kk + sel + 1][wn + 16 + l15];
            acc[0][0] = __builtin_amdgcn_wmma_f32_16x16x4_f32(false, a0, false, b0, (short)0, acc[0][0], false, false);
            acc[0][1] = __builtin_amdgcn_wmma_f32_16x16x4_f32(false, a0, false, b1, (short)0, acc[0][1], false, false);
            acc[1][0] = __builtin_amdgcn_wmma_f32_16x16x4_f32(false, a1, false, b0, (short)0, acc[1][0], false, false);
            acc[1][1] = __builtin_amdgcn_wmma_f32_16x16x4_f32(false, a1, false, b1, (short)0, acc[1][1], false, false);
#else
            // Layout-identical scalar fallback (same C fragment mapping).
            const int mrow = (lane < 16) ? 0 : 8;
            #pragma unroll
            for (int mt = 0; mt < 2; ++mt)
                #pragma unroll
                for (int nt = 0; nt < 2; ++nt)
                    #pragma unroll
                    for (int r = 0; r < 8; ++r) {
                        float s = 0.0f;
                        #pragma unroll
                        for (int k4 = 0; k4 < 4; ++k4)
                            s += Am[kk + k4][wm + mt * 16 + mrow + r] *
                                 Bs[kk + k4][wn + nt * 16 + l15];
                        acc[mt][nt][r] += s;
                    }
#endif
        }
    }

    const int mrow = (lane < 16) ? 0 : 8;
    #pragma unroll
    for (int mt = 0; mt < 2; ++mt)
        #pragma unroll
        for (int nt = 0; nt < 2; ++nt)
            #pragma unroll
            for (int r = 0; r < 8; ++r) {
                int row = m0 + wm + mt * 16 + mrow + r;
                int col = n0 + wn + nt * 16 + l15;
                float v = acc[mt][nt][r];
                if (UPDATE)
                    Out[(long)row * 1024 + col] =
                        1.5f * Worig[(long)row * 1024 + col] - 0.5f * v;
                else
                    Out[(long)row * 1024 + col] = v;
            }
}

// ---------------------------------------------------------------------------
// Phase 2: convert final W (fp32 [i][o]) -> transposed bf16 WT[o][i]
// ---------------------------------------------------------------------------
__global__ void convert_transpose_bf16(const float* __restrict__ W,
                                       unsigned short* __restrict__ WT) {
    int idx = blockIdx.x * blockDim.x + threadIdx.x;  // linear over output
    int o = idx >> 10;
    int i = idx & 1023;
    WT[(long)o * 1024 + i] = f2bf_rne(W[(long)i * 1024 + o]);
}

// ---------------------------------------------------------------------------
// Phase 3: Y[m][n] = sum_k bf16(X[m][k]) * Wbf[k][n] + bias[n]
// 128x128 tile per 256-thread block (8 waves as 4x2; wave = 32x64 = 2x4 tiles)
// Double-buffered LDS pipeline:
//   - B operand (bf16 copy)   : global_load_async_to_lds_b128  (ASYNCcnt)
//   - A operand (fp32->bf16)  : register prefetch + convert + ds_store
// 16 K-chunks of 64; two K=32 bf16 WMMA steps per chunk (16 WMMA / chunk).
// ---------------------------------------------------------------------------
__global__ __launch_bounds__(256) void main_gemm(
        const float* __restrict__ X,
        const unsigned short* __restrict__ WT,   // bf16 bits, [o][i]
        const float* __restrict__ bias,
        float* __restrict__ Y) {
    __shared__ unsigned short As[2][128][64];   // As[buf][m][k]  bf16
    __shared__ unsigned short Bt[2][128][64];   // Bt[buf][n][k]  bf16 (N-major)
    const int tid  = threadIdx.x;
    const int lane = tid & 31;
    const int wid  = tid >> 5;
    const int m0 = blockIdx.y * 128;
    const int n0 = blockIdx.x * 128;
    const int wm = (wid >> 1) * 32;   // 0,32,64,96
    const int wn = (wid & 1) * 64;    // 0,64
    const int l15   = lane & 15;
    const int sel8  = (lane < 16) ? 0 : 8;    // A operand lane-half K offset
    const int sel16 = sel8 * 2;               // unused (B operand also uses sel8)

    // Per-thread staging coordinates
    const int ar  = tid >> 4;              // A: rows tid>>4 + 16*i? no: idx-based below
    (void)ar;

    v8f acc[2][4] = {};

    // ---- prologue: stage chunk 0 into buffer 0 ----
    #pragma unroll
    for (int i = 0; i < 4; ++i) {          // B: 1024 x 16B
        int idx = tid + i * 256;
        int r   = idx >> 3;
        int c8  = (idx & 7) * 8;
        async_copy_b128(WT + (long)(n0 + r) * 1024 + c8, &Bt[0][r][c8]);
    }
    #pragma unroll
    for (int i = 0; i < 8; ++i) {          // A: 2048 float4, convert to bf16
        int idx = tid + i * 256;
        int r   = idx >> 4;
        int c4  = (idx & 15) * 4;
        float4 v = *(const float4*)(X + (long)(m0 + r) * 1024 + c4);
        ushort4 s;
        s.x = f2bf_rne(v.x); s.y = f2bf_rne(v.y);
        s.z = f2bf_rne(v.z); s.w = f2bf_rne(v.w);
        *(ushort4*)&As[0][r][c4] = s;
    }
#if USE_ASYNC_B
    wait_async0();
#endif
    __syncthreads();

    // ---- main pipeline ----
    for (int c = 0; c < 16; ++c) {
        const int cur = c & 1;
        const int nxt = cur ^ 1;
        const int k0n = (c + 1) * 64;

        // Issue next chunk's loads first (overlap with WMMA below).
        float4 pre[8];
        if (c < 15) {
            #pragma unroll
            for (int i = 0; i < 4; ++i) {
                int idx = tid + i * 256;
                int r   = idx >> 3;
                int c8  = (idx & 7) * 8;
                async_copy_b128(WT + (long)(n0 + r) * 1024 + k0n + c8,
                                &Bt[nxt][r][c8]);
            }
            #pragma unroll
            for (int i = 0; i < 8; ++i) {
                int idx = tid + i * 256;
                int r   = idx >> 4;
                int c4  = (idx & 15) * 4;
                pre[i] = *(const float4*)(X + (long)(m0 + r) * 1024 + k0n + c4);
            }
        }

        // Compute current chunk: 2 x (K=32) steps, 8 WMMA each.
        #pragma unroll
        for (int kcs = 0; kcs < 2; ++kcs) {
            const int kc = kcs * 32;
            v16bf a[2], b[4];
            #pragma unroll
            for (int mt = 0; mt < 2; ++mt) {
                const unsigned short* p = &As[cur][wm + mt * 16 + l15][kc + sel8];
                v8bf lo = *(const v8bf*)p;          // K = kc+sel8    .. +7
                v8bf hi = *(const v8bf*)(p + 16);   // K = kc+16+sel8 .. +7
                a[mt] = __builtin_shufflevector(lo, hi,
                        0,1,2,3,4,5,6,7,8,9,10,11,12,13,14,15);
            }
            #pragma unroll
            for (int nt = 0; nt < 4; ++nt) {
                const unsigned short* p = &Bt[cur][wn + nt * 16 + l15][kc + sel8];
                v8bf lo = *(const v8bf*)p;          // K = kc+sel8    .. +7
                v8bf hi = *(const v8bf*)(p + 16);   // K = kc+16+sel8 .. +7
                b[nt] = __builtin_shufflevector(lo, hi,
                        0,1,2,3,4,5,6,7,8,9,10,11,12,13,14,15);
            }
            #pragma unroll
            for (int mt = 0; mt < 2; ++mt)
                #pragma unroll
                for (int nt = 0; nt < 4; ++nt)
                    acc[mt][nt] = __builtin_amdgcn_wmma_f32_16x16x32_bf16(
                        false, a[mt], false, b[nt], (short)0, acc[mt][nt],
                        false, false);
        }

        // Drain staging for next chunk, then rendezvous.
        if (c < 15) {
            #pragma unroll
            for (int i = 0; i < 8; ++i) {
                int idx = tid + i * 256;
                int r   = idx >> 4;
                int c4  = (idx & 15) * 4;
                ushort4 s;
                s.x = f2bf_rne(pre[i].x); s.y = f2bf_rne(pre[i].y);
                s.z = f2bf_rne(pre[i].z); s.w = f2bf_rne(pre[i].w);
                *(ushort4*)&As[nxt][r][c4] = s;
            }
#if USE_ASYNC_B
            wait_async0();
#endif
        }
        __syncthreads();
    }

    // ---- epilogue ----
    const int mrow = (lane < 16) ? 0 : 8;
    #pragma unroll
    for (int nt = 0; nt < 4; ++nt) {
        int col = n0 + wn + nt * 16 + l15;
        float bv = bias[col];
        #pragma unroll
        for (int mt = 0; mt < 2; ++mt)
            #pragma unroll
            for (int r = 0; r < 8; ++r) {
                int row = m0 + wm + mt * 16 + mrow + r;
                Y[(long)row * 1024 + col] = acc[mt][nt][r] + bv;
            }
    }
}

// ---------------------------------------------------------------------------
extern "C" void kernel_launch(void* const* d_in, const int* in_sizes, int n_in,
                              void* d_out, int out_size, void* d_ws, size_t ws_size,
                              hipStream_t stream) {
    const float* x      = (const float*)d_in[0];   // [4,8192,1024] = [32768][1024]
    const float* kernel = (const float*)d_in[1];   // [1024][1024]
    const float* bias   = (const float*)d_in[2];   // [1024]
    float* y = (float*)d_out;

    char* ws = (char*)d_ws;
    float*          W0   = (float*)(ws);                       // 4 MB
    float*          W1   = (float*)(ws + (4u << 20));          // 4 MB
    float*          T    = (float*)(ws + (8u << 20));          // 4 MB
    unsigned short* WbfT = (unsigned short*)(ws + (12u << 20));// 2 MB

    // W0 = kernel / 1024
    scale_w<<<1048576 / 256, 256, 0, stream>>>(kernel, W0);

    // 20 Bjorck iterations: T = Wa^T Wa ; Wb = 1.5 Wa - 0.5 Wa T
    float* Wa = W0;
    float* Wb = W1;
    for (int it = 0; it < 20; ++it) {
        bjorck_gemm<false, false><<<dim3(16, 16), 128, 0, stream>>>(Wa, Wa, Wa, T);
        bjorck_gemm<true,  true ><<<dim3(16, 16), 128, 0, stream>>>(Wa, T, Wa, Wb);
        float* tmp = Wa; Wa = Wb; Wb = tmp;
    }

    // WbfT[o][i] = bf16(Wa[i][o])
    convert_transpose_bf16<<<1048576 / 256, 256, 0, stream>>>(Wa, WbfT);

    // Y = X * W + bias  (bf16 WMMA, fp32 accumulate, async+double-buffered)
    main_gemm<<<dim3(8, 256), 256, 0, stream>>>(x, WbfT, bias, y);
}
